// SpatialEncoderLayer_618475290644
// MI455X (gfx1250) — matrix operand order repaired and checked by reading, not compiled, not run on hardware
//
#include <hip/hip_runtime.h>
#include <hip/hip_bf16.h>

// ---------------------------------------------------------------------------
// SpatialEncoderLayer for MI455X (gfx1250): fused per-frame graph-attention
// encoder layer. GEMMs via v_wmma_f32_16x16x32_bf16 (wave32 WMMA); the x tile
// is staged into LDS by the Tensor Data Mover (tensor_load_to_lds + TENSORcnt).
// F=4096 frames, N=64 nodes, C=256 channels, H=8 heads, D=32.
// ---------------------------------------------------------------------------

typedef __bf16 bf16_t;
typedef __attribute__((ext_vector_type(16))) __bf16 v16bf;
typedef __attribute__((ext_vector_type(8)))  float  v8f;
typedef unsigned int u32x4 __attribute__((ext_vector_type(4)));
typedef int          i32x4 __attribute__((ext_vector_type(4)));
typedef int          i32x8 __attribute__((ext_vector_type(8)));

#define N_NODES 64
#define C_DIM   256
#define H_HEADS 8
#define D_HEAD  32
#define EPS_LN  1e-5f

// ---- LDS layout (bytes), dynamic shared memory, total 258048 <= 320KB -----
#define XS 264   // bf16 row stride for x / x1 (64 rows)
#define QS 264   // bf16 row stride for q (64 rows)
#define KS 264   // bf16 row stride for k (64 rows)
#define VS 72    // bf16 row stride for v^T (256 rows of 64)
#define SCS 68   // f32 row stride for per-wave score rows (16 rows/wave)
#define PBS 72   // bf16 row stride for per-wave prob rows (16 rows/wave)
#define TS 260   // f32 row stride for x(f32)/t/x1f (64 rows)  [TDM pad target]
#define FS 260   // f32 row stride for ffn-out buffer (reuses k/vT region)

#define OFF_XB  0
#define OFF_QB  (OFF_XB + 64 * XS * 2)          // 33792
#define OFF_KB  (OFF_QB + 64 * QS * 2)          // 67584
#define OFF_VT  (OFF_KB + 64 * KS * 2)          // 101376
#define OFF_SC  (OFF_VT + 256 * VS * 2)         // 138240
#define OFF_PB  (OFF_SC + 8 * 16 * SCS * 4)     // 173056
#define OFF_TF  (OFF_PB + 8 * 16 * PBS * 2)     // 191488
#define SMEM_BYTES (OFF_TF + 64 * TS * 4)       // 258048
#define OFF_FB  OFF_KB                          // phase-E f32 buffer reuses k+vT

// ---------------------------------------------------------------------------
// Fragment loaders: contiguous k order k = 16*(lane/16) + s, same for A and B.
// ---------------------------------------------------------------------------
__device__ __forceinline__ v16bf frag_a(const bf16_t* base, int ld, int lane) {
    const int r  = lane & 15;
    const int kb = (lane >> 4) * 16;
    const bf16_t* p = base + r * ld + kb;
    v16bf a;
#pragma unroll
    for (int i = 0; i < 8; ++i) a[i] = p[i];
#pragma unroll
    for (int i = 0; i < 8; ++i) a[8 + i] = p[8 + i];
    return a;
}

__device__ __forceinline__ v16bf frag_b(const bf16_t* bt, int ld, int lane) {
    const int n  = lane & 15;
    const int kb = (lane >> 4) * 16;
    const bf16_t* p = bt + n * ld + kb;
    v16bf b;
#pragma unroll
    for (int i = 0; i < 16; ++i) b[i] = p[i];
    return b;
}

__device__ __forceinline__ v8f wmma_bf16(v16bf a, v16bf b, v8f c) {
    return __builtin_amdgcn_wmma_f32_16x16x32_bf16(
        false, a, false, b, (short)0, c, false, false);
}

// ---------------------------------------------------------------------------
// Prep: convert weights to bf16, transposed to [out][in] for B-fragments.
// ---------------------------------------------------------------------------
__global__ void prep_weights(const float* __restrict__ wqkv,
                             const float* __restrict__ w1,
                             const float* __restrict__ w2,
                             bf16_t* __restrict__ wqkvT,
                             bf16_t* __restrict__ w1T,
                             bf16_t* __restrict__ w2T) {
    int i = blockIdx.x * blockDim.x + threadIdx.x;
    if (i < 768 * 256) {
        int n = i >> 8, k = i & 255;
        wqkvT[i] = (bf16_t)wqkv[k * 768 + n];
    } else if (i < 768 * 256 + 65536) {
        int j = i - 768 * 256;
        int n = j >> 8, k = j & 255;
        w1T[j] = (bf16_t)w1[k * 256 + n];
    } else if (i < 768 * 256 + 2 * 65536) {
        int j = i - 768 * 256 - 65536;
        int n = j >> 8, k = j & 255;
        w2T[j] = (bf16_t)w2[k * 256 + n];
    }
}

// ---------------------------------------------------------------------------
// Main fused kernel: one block = one frame.
// ---------------------------------------------------------------------------
__global__ __launch_bounds__(256, 1)
void encoder_layer_kernel(const float* __restrict__ x,
                          const int*   __restrict__ adj,
                          const bf16_t* __restrict__ wqkvT,
                          const bf16_t* __restrict__ w1T,
                          const bf16_t* __restrict__ w2T,
                          const float* __restrict__ b1,
                          const float* __restrict__ b2,
                          const float* __restrict__ ln1g,
                          const float* __restrict__ ln1b,
                          const float* __restrict__ ln2g,
                          const float* __restrict__ ln2b,
                          float* __restrict__ out) {
    extern __shared__ char smem[];
    bf16_t* xbf  = (bf16_t*)(smem + OFF_XB);  // x, later x1 (bf16)
    bf16_t* qbf  = (bf16_t*)(smem + OFF_QB);  // q (scaled), later ffn h
    bf16_t* kbf  = (bf16_t*)(smem + OFF_KB);  // k
    bf16_t* vtbf = (bf16_t*)(smem + OFF_VT);  // v^T [c][node]
    float*  scA  = (float*) (smem + OFF_SC);  // per-wave score rows
    bf16_t* pbA  = (bf16_t*)(smem + OFF_PB);  // per-wave prob rows
    float*  tf   = (float*) (smem + OFF_TF);  // x(f32) via TDM, t, then x1(f32)
    float*  fb   = (float*) (smem + OFF_FB);  // phase-E f32 (reuses k/vT)

    const int frame  = blockIdx.x;
    const int tid    = threadIdx.x;
    const int waveId = tid >> 5;
    const int lane   = tid & 31;
    const int col    = lane & 15;
    const int rb     = (lane >> 4) * 8;       // C/D row base per half-wave
    const float* xg  = x + (size_t)frame * N_NODES * C_DIM;

    // ---- TDM: DMA x (64x256 f32) into tf with row stride 260 floats -------
    // D# per CDNA5 ISA ch.8: 2D tensor, data_size=4B, tile 256x64,
    // pad_interval=256 DWORDs (enc 7), pad_amount=4 DWORDs (enc 3).
    if (waveId == 0) {
        const unsigned long long ga = (unsigned long long)(uintptr_t)xg;
        u32x4 g0;
        g0[0] = 1u;                                   // count=1, user mode
        g0[1] = (unsigned)OFF_TF;                     // lds_addr (dyn LDS base=0)
        g0[2] = (unsigned)(ga & 0xFFFFFFFFu);         // global_addr[31:0]
        g0[3] = (unsigned)((ga >> 32) & 0x1FFFFFFu)   // global_addr[56:32]
              | (2u << 30);                           // type=2 (image)
        i32x8 g1;
        g1[0] = (2 << 16) | (1 << 20) | (7 << 22) | (3 << 25);
        g1[1] = (int)(256u << 16);                    // tensor_dim0 = 256
        g1[2] = (int)(64u << 16);                     // tensor_dim1 = 64
        g1[3] = (int)(256u << 16);                    // tile_dim0 = 256
        g1[4] = 64;                                   // tile_dim1 = 64
        g1[5] = 256;                                  // tensor_dim0_stride = 256
        g1[6] = 0; g1[7] = 0;
        i32x4 gz = {0, 0, 0, 0};
#if defined(__clang_major__) && (__clang_major__ >= 23)
        i32x8 gz8 = {0, 0, 0, 0, 0, 0, 0, 0};
        __builtin_amdgcn_tensor_load_to_lds(g0, g1, gz, gz, gz8, 0);
#else
        __builtin_amdgcn_tensor_load_to_lds(g0, g1, gz, gz, 0);
#endif
        __builtin_amdgcn_s_wait_tensorcnt(0);
    }
    __syncthreads();

    // ---- convert staged x f32 -> bf16 ------------------------------------
    for (int i = tid; i < N_NODES * C_DIM; i += 256) {
        int r = i >> 8, c = i & 255;
        xbf[r * XS + c] = (bf16_t)tf[r * TS + c];
    }
    __syncthreads();

    // ---- Phase A: qkv = x @ Wqkv; hoist A-frags per 16-row block ----------
    const float qscale = 0.17677669529663688f;  // 1/sqrt(D=32)
    for (int mb = 0; mb < 4; ++mb) {
        const int m0 = mb * 16;
        v16bf a[8];
#pragma unroll
        for (int kk = 0; kk < 8; ++kk)
            a[kk] = frag_a(&xbf[m0 * XS + kk * 32], XS, lane);
        for (int nt = waveId; nt < 48; nt += 8) {
            const int n0 = nt * 16;
            v8f acc = {};
#pragma unroll
            for (int kk = 0; kk < 8; ++kk)
                acc = wmma_bf16(a[kk], frag_b(&wqkvT[n0 * 256 + kk * 32], 256, lane), acc);
            if (n0 < 256) {
#pragma unroll
                for (int i = 0; i < 8; ++i)
                    qbf[(m0 + rb + i) * QS + n0 + col] = (bf16_t)(acc[i] * qscale);
            } else if (n0 < 512) {
#pragma unroll
                for (int i = 0; i < 8; ++i)
                    kbf[(m0 + rb + i) * KS + (n0 - 256) + col] = (bf16_t)acc[i];
            } else {
#pragma unroll
                for (int i = 0; i < 8; ++i)
                    vtbf[((n0 - 512) + col) * VS + (m0 + rb + i)] = (bf16_t)acc[i];
            }
        }
    }
    __syncthreads();

    // ---- Phase B: one head per wave --------------------------------------
    const int head = waveId;
    const int hoff = head * D_HEAD;
    float*  scf = scA + waveId * 16 * SCS;
    bf16_t* pbw = pbA + waveId * 16 * PBS;

    for (int mb = 0; mb < 4; ++mb) {
        const int m0 = mb * 16;
        // scores: 4 tiles of 16x16, K = D = 32 -> single WMMA each
        v16bf aq = frag_a(&qbf[m0 * QS + hoff], QS, lane);
#pragma unroll
        for (int nt = 0; nt < 4; ++nt) {
            const int n0 = nt * 16;
            v16bf bk = frag_b(&kbf[n0 * KS + hoff], KS, lane);
            v8f s = {};
            s = wmma_bf16(aq, bk, s);
#pragma unroll
            for (int i = 0; i < 8; ++i)
                scf[(rb + i) * SCS + n0 + col] = s[i];
        }
        // masked softmax: lanes 0..15 each own one query row
        if (lane < 16) {
            const int r = lane;
            float* srow = scf + r * SCS;
            const int* arow = adj + (m0 + r) * N_NODES;
            unsigned long long mrow = 0ull;
            float mx = -3.0e38f;
            for (int c = 0; c < 64; ++c) {
                if (arow[c] > 0) {
                    mrow |= 1ull << c;
                    mx = fmaxf(mx, srow[c]);
                }
            }
            float sum = 0.f;
            for (int c = 0; c < 64; ++c) {
                float e = ((mrow >> c) & 1ull) ? __expf(srow[c] - mx) : 0.f;
                srow[c] = e;
                sum += e;
            }
            const float inv = 1.f / sum;
            bf16_t* prow = pbw + r * PBS;
            for (int c = 0; c < 64; ++c) prow[c] = (bf16_t)(srow[c] * inv);
        }
        __syncthreads();  // uniform; publish probs to whole wave

        // t_block(16x32) = P(16x64) @ V_head(64x32)
#pragma unroll
        for (int dt = 0; dt < 2; ++dt) {
            const int d0 = dt * 16;
            v8f acc = {};
#pragma unroll
            for (int k0 = 0; k0 < 64; k0 += 32) {
                v16bf a = frag_a(&pbw[k0], PBS, lane);
                v16bf b = frag_b(&vtbf[(hoff + d0) * VS + k0], VS, lane);
                acc = wmma_bf16(a, b, acc);
            }
#pragma unroll
            for (int i = 0; i < 8; ++i)
                tf[(m0 + rb + i) * TS + hoff + d0 + col] = acc[i];
        }
    }
    __syncthreads();

    // ---- Phase C: x1 = LN1(x + t); keep f32 in tf, bf16 in xbf -----------
    if (tid < N_NODES) {
        const int node = tid;
        float* trow = tf + node * TS;
        const float* xr = xg + node * C_DIM;
        float sum = 0.f, sumsq = 0.f;
        for (int c = 0; c < C_DIM; ++c) {
            float v = xr[c] + trow[c];
            trow[c] = v;
            sum += v;
            sumsq += v * v;
        }
        const float mu = sum * (1.0f / C_DIM);
        const float var = sumsq * (1.0f / C_DIM) - mu * mu;
        const float rs = rsqrtf(var + EPS_LN);
        for (int c = 0; c < C_DIM; ++c) {
            float o = (trow[c] - mu) * rs * ln1g[c] + ln1b[c];
            trow[c] = o;
            xbf[node * XS + c] = (bf16_t)o;
        }
    }
    __syncthreads();

    // ---- Phase D: h = relu(x1 @ W1 + b1) -> qbf (bf16) --------------------
    for (int mb = 0; mb < 4; ++mb) {
        const int m0 = mb * 16;
        v16bf a[8];
#pragma unroll
        for (int kk = 0; kk < 8; ++kk)
            a[kk] = frag_a(&xbf[m0 * XS + kk * 32], XS, lane);
        for (int nt = waveId; nt < 16; nt += 8) {
            const int n0 = nt * 16;
            v8f acc = {};
#pragma unroll
            for (int kk = 0; kk < 8; ++kk)
                acc = wmma_bf16(a[kk], frag_b(&w1T[n0 * 256 + kk * 32], 256, lane), acc);
            const float bias = b1[n0 + col];
#pragma unroll
            for (int i = 0; i < 8; ++i)
                qbf[(m0 + rb + i) * QS + n0 + col] = (bf16_t)fmaxf(acc[i] + bias, 0.f);
        }
    }
    __syncthreads();

    // ---- Phase E: v2 = x1 + (h @ W2 + b2) -> fb (f32) ---------------------
    for (int mb = 0; mb < 4; ++mb) {
        const int m0 = mb * 16;
        v16bf a[8];
#pragma unroll
        for (int kk = 0; kk < 8; ++kk)
            a[kk] = frag_a(&qbf[m0 * QS + kk * 32], QS, lane);
        for (int nt = waveId; nt < 16; nt += 8) {
            const int n0 = nt * 16;
            v8f acc = {};
#pragma unroll
            for (int kk = 0; kk < 8; ++kk)
                acc = wmma_bf16(a[kk], frag_b(&w2T[n0 * 256 + kk * 32], 256, lane), acc);
            const float bias = b2[n0 + col];
#pragma unroll
            for (int i = 0; i < 8; ++i) {
                const int r = m0 + rb + i;
                fb[r * FS + n0 + col] = acc[i] + bias + tf[r * TS + n0 + col];
            }
        }
    }
    __syncthreads();

    // ---- LN2 -> global output (f32) --------------------------------------
    if (tid < N_NODES) {
        const int node = tid;
        const float* vrow = fb + node * FS;
        float sum = 0.f, sumsq = 0.f;
        for (int c = 0; c < C_DIM; ++c) {
            float v = vrow[c];
            sum += v;
            sumsq += v * v;
        }
        const float mu = sum * (1.0f / C_DIM);
        const float var = sumsq * (1.0f / C_DIM) - mu * mu;
        const float rs = rsqrtf(var + EPS_LN);
        float* orow = out + ((size_t)frame * N_NODES + node) * C_DIM;
        for (int c = 0; c < C_DIM; ++c)
            orow[c] = (vrow[c] - mu) * rs * ln2g[c] + ln2b[c];
    }
}

// ---------------------------------------------------------------------------
extern "C" void kernel_launch(void* const* d_in, const int* in_sizes, int n_in,
                              void* d_out, int out_size, void* d_ws, size_t ws_size,
                              hipStream_t stream) {
    const float* x    = (const float*)d_in[0];
    const int*   adj  = (const int*)  d_in[1];
    const float* wqkv = (const float*)d_in[2];
    const float* ln1g = (const float*)d_in[3];
    const float* ln1b = (const float*)d_in[4];
    const float* ln2g = (const float*)d_in[5];
    const float* ln2b = (const float*)d_in[6];
    const float* w1   = (const float*)d_in[7];
    const float* b1   = (const float*)d_in[8];
    const float* w2   = (const float*)d_in[9];
    const float* b2   = (const float*)d_in[10];
    float* out = (float*)d_out;

    const int F = in_sizes[0] / (N_NODES * C_DIM);

    bf16_t* wqkvT = (bf16_t*)d_ws;                          // 768*256 bf16
    bf16_t* w1T   = (bf16_t*)((char*)d_ws + 768 * 256 * 2); // 256*256 bf16
    bf16_t* w2T   = (bf16_t*)((char*)d_ws + 768 * 256 * 2 + 65536 * 2);

    const int prep_elems = 768 * 256 + 2 * 65536;
    prep_weights<<<(prep_elems + 255) / 256, 256, 0, stream>>>(
        wqkv, w1, w2, wqkvT, w1T, w2T);

    (void)hipFuncSetAttribute((const void*)encoder_layer_kernel,
                              hipFuncAttributeMaxDynamicSharedMemorySize,
                              SMEM_BYTES);

    encoder_layer_kernel<<<F, 256, SMEM_BYTES, stream>>>(
        x, adj, wqkvT, w1T, w2T, b1, b2, ln1g, ln1b, ln2g, ln2b, out);
}
